// DecoderRNN_74552042324372
// MI455X (gfx1250) — compile-verified, matrix operand order checked by reading, not run on hardware
//
#include <hip/hip_runtime.h>
#include <math.h>

// ---------------------------------------------------------------------------
// DecoderRNN step for MI455X (gfx1250, wave32, WMMA).
// B=128, E=H=1024, V=32001. HBM-bound on streaming w_out (131 MB); every
// weight element is read exactly once, activations (256 KB bf16) stay in L2.
// GEMMs use v_wmma_f32_16x16x32_bf16 (f32 accumulate), one wave per 16-col
// N-stripe computing the full M=128 so the weight fragment is reused 8x.
// Tail lanes (V=32001) read from a zeroed scratch row instead of predicating
// inside the hot loop; f32->bf16 goes through __builtin_convertvector so the
// backend can use packed v_cvt_pk_bf16_f32.
// ---------------------------------------------------------------------------

#define BB   128
#define EE   1024
#define HH   1024
#define VV   32001

typedef __attribute__((ext_vector_type(16))) __bf16 v16bf;
typedef __attribute__((ext_vector_type(8)))  __bf16 v8bf;
typedef __attribute__((ext_vector_type(16))) float  v16f;
typedef __attribute__((ext_vector_type(8)))  float  v8f;

static __device__ __forceinline__ __bf16 to_bf16(float f) { return (__bf16)f; }

// ---- embedding gather + ReLU -> bf16 --------------------------------------
__global__ void embed_relu_bf16(const int* __restrict__ idx,
                                const float* __restrict__ emb,
                                __bf16* __restrict__ x) {
    int b = blockIdx.x;
    const float* src = emb + (size_t)idx[b] * EE;
    __bf16* dst = x + (size_t)b * EE;
    for (int e = threadIdx.x; e < EE; e += blockDim.x) {
        float v = src[e];
        v = v > 0.0f ? v : 0.0f;
        dst[e] = to_bf16(v);
    }
}

// ---- f32 -> bf16 bulk convert ---------------------------------------------
__global__ void f32_to_bf16(const float* __restrict__ in,
                            __bf16* __restrict__ out, int n) {
    int i = blockIdx.x * blockDim.x + threadIdx.x;
    if (i < n) out[i] = to_bf16(in[i]);
}

// ---- zero a small f32 region (scratch row for tail lanes) ------------------
__global__ void zero_f32(float* __restrict__ p, int n) {
    int i = blockIdx.x * blockDim.x + threadIdx.x;
    if (i < n) p[i] = 0.0f;
}

// ---- G[M=128,N] = Xbf16[128,K] * Wf32[N,K]^T + bias[N] ----------------------
// One wave (32 threads) per 16-column N tile; computes full M=128.
// Fragment layout per ISA 7.12.2 (16-bit A/B, 16x16x32):
//   lane l: half g = l>>4, idx = l&15; vector slots 0..7  hold K = k0+8g..+7
//                                       vector slots 8..15 hold K = k0+16+8g..+7
// A: idx = M row; B: idx = N column (B = W^T, loaded from W rows).
// Invalid tail lanes (n >= N) stream from `zrow` (K zeros) so the K-loop has
// no per-lane predication and EXEC stays all-ones for every WMMA.
__global__ void __launch_bounds__(32)
gemm_xwT_bias_wmma(const __bf16* __restrict__ X,
                   const float*  __restrict__ W,
                   const float*  __restrict__ bias,
                   const float*  __restrict__ zrow,
                   float*        __restrict__ G,
                   int K, int N, int ldG) {
    const int lane = threadIdx.x;          // 0..31
    const int nl   = lane & 15;
    const int g    = lane >> 4;
    const int n0   = blockIdx.x * 16;
    const int n    = n0 + nl;
    const bool nvalid = (n < N);           // only false in the V=32001 tail tile

    const float* wrow = nvalid ? (W + (size_t)n * K) : zrow;

    v8f acc[8] = {};                       // 8 M-tiles x (16x16 f32 tile / wave)

    for (int k0 = 0; k0 < K; k0 += 32) {
        // ---- B fragment: 16 weights of row n, packed f32 -> bf16 ----------
        const float* p0 = wrow + k0 + g * 8;        // slots 0..7
        const float* p1 = wrow + k0 + 16 + g * 8;   // slots 8..15
        if (k0 + 64 < K) __builtin_prefetch(p0 + 64, 0, 0);  // global_prefetch_b8
        v8f r0 = *(const v8f*)(p0);
        v8f r1 = *(const v8f*)(p1);
        v16f wf = __builtin_shufflevector(r0, r1, 0, 1, 2, 3, 4, 5, 6, 7,
                                          8, 9, 10, 11, 12, 13, 14, 15);
        v16bf bfrag = __builtin_convertvector(wf, v16bf);

        // ---- reuse B fragment across all 8 M tiles ------------------------
#pragma unroll
        for (int mt = 0; mt < 8; ++mt) {
            const __bf16* xp = X + (size_t)(mt * 16 + nl) * K + k0 + g * 8;
            v8bf a0 = *(const v8bf*)(xp);        // K = k0+8g    .. +7
            v8bf a1 = *(const v8bf*)(xp + 16);   // K = k0+16+8g .. +7
            v16bf afrag = __builtin_shufflevector(a0, a1, 0, 1, 2, 3, 4, 5, 6, 7,
                                                  8, 9, 10, 11, 12, 13, 14, 15);
            acc[mt] = __builtin_amdgcn_wmma_f32_16x16x32_bf16(
                false, afrag, false, bfrag, (short)0, acc[mt], false, false);
        }
    }

    // ---- epilogue: bias + store (C/D layout: lane half selects M+8) -------
    const float bn = nvalid ? bias[n] : 0.0f;
#pragma unroll
    for (int mt = 0; mt < 8; ++mt) {
#pragma unroll
        for (int j = 0; j < 8; ++j) {
            int m = mt * 16 + j + 8 * g;
            if (nvalid) G[(size_t)m * ldG + n] = acc[mt][j] + bn;
        }
    }
}

// ---- GRU gate math: h = (1-z)*n + z*h_prev ---------------------------------
__global__ void gru_pointwise(const float* __restrict__ gi,
                              const float* __restrict__ gh,
                              const float* __restrict__ hprev,
                              float* __restrict__ hout,
                              __bf16* __restrict__ houtb) {
    int i = blockIdx.x * blockDim.x + threadIdx.x;   // 0 .. B*H-1
    int b = i >> 10;                                  // H == 1024
    int j = i & 1023;
    const float* gib = gi + (size_t)b * (3 * HH);
    const float* ghb = gh + (size_t)b * (3 * HH);
    float ir = gib[j],          hr = ghb[j];
    float iz = gib[HH + j],     hz = ghb[HH + j];
    float in = gib[2 * HH + j], hn = ghb[2 * HH + j];
    float r = 1.0f / (1.0f + __expf(-(ir + hr)));
    float z = 1.0f / (1.0f + __expf(-(iz + hz)));
    float nn = tanhf(in + r * hn);
    float h = (1.0f - z) * nn + z * hprev[i];
    hout[i]  = h;
    houtb[i] = to_bf16(h);
}

// ---- in-place log_softmax over each row of [B, V] --------------------------
__global__ void log_softmax_inplace(float* __restrict__ out) {
    __shared__ float red[256];
    float* row = out + (size_t)blockIdx.x * VV;
    int t = threadIdx.x;

    float m = -3.0e38f;
    for (int v = t; v < VV; v += 256) m = fmaxf(m, row[v]);
    red[t] = m; __syncthreads();
    for (int s = 128; s > 0; s >>= 1) {
        if (t < s) red[t] = fmaxf(red[t], red[t + s]);
        __syncthreads();
    }
    m = red[0]; __syncthreads();

    float sum = 0.0f;
    for (int v = t; v < VV; v += 256) sum += __expf(row[v] - m);
    red[t] = sum; __syncthreads();
    for (int s = 128; s > 0; s >>= 1) {
        if (t < s) red[t] += red[t + s];
        __syncthreads();
    }
    float lse = m + __logf(red[0]);
    for (int v = t; v < VV; v += 256) row[v] -= lse;
}

// ---------------------------------------------------------------------------
extern "C" void kernel_launch(void* const* d_in, const int* in_sizes, int n_in,
                              void* d_out, int out_size, void* d_ws, size_t ws_size,
                              hipStream_t stream) {
    const int*   idx    = (const int*)  d_in[0];
    const float* hidden = (const float*)d_in[1];   // [2, B, H]
    const float* emb    = (const float*)d_in[2];   // [V, E]
    const float* w_ih0  = (const float*)d_in[3];   // [3H, E]
    const float* w_hh0  = (const float*)d_in[4];   // [3H, H]
    const float* b_ih0  = (const float*)d_in[5];
    const float* b_hh0  = (const float*)d_in[6];
    const float* w_ih1  = (const float*)d_in[7];   // [3H, H]
    const float* w_hh1  = (const float*)d_in[8];
    const float* b_ih1  = (const float*)d_in[9];
    const float* b_hh1  = (const float*)d_in[10];
    const float* w_out  = (const float*)d_in[11];  // [V, H]
    const float* b_out  = (const float*)d_in[12];

    float* out = (float*)d_out;                    // [B*V] logits then [2*B*H]
    float* out_h0 = out + (size_t)BB * VV;
    float* out_h1 = out_h0 + (size_t)BB * HH;

    // workspace (all offsets 256B aligned)
    char* ws = (char*)d_ws;
    __bf16* x_bf     = (__bf16*)(ws + 0);           //  B*E bf16   (256 KB)
    __bf16* hprev_bf = (__bf16*)(ws + 262144);      // 2*B*H bf16  (512 KB)
    __bf16* h0_bf    = (__bf16*)(ws + 786432);      //  B*H bf16
    __bf16* h1_bf    = (__bf16*)(ws + 1048576);     //  B*H bf16
    float*  gi       = (float*) (ws + 1310720);     //  B*3H f32   (1.5 MB)
    float*  gh       = (float*) (ws + 2883584);     //  B*3H f32   (1.5 MB)
    float*  zrow     = (float*) (ws + 4456448);     //  1024 f32 of zeros (4 KB)

    // 0) zero scratch row for tail lanes (deterministic every call)
    zero_f32<<<4, 256, 0, stream>>>(zrow, 1024);

    // 1) x = relu(emb[idx]) -> bf16 ; hidden -> bf16
    embed_relu_bf16<<<BB, 256, 0, stream>>>(idx, emb, x_bf);
    f32_to_bf16<<<(2 * BB * HH) / 256, 256, 0, stream>>>(hidden, hprev_bf, 2 * BB * HH);

    // 2) layer 0: gi = x @ w_ih0^T + b_ih0 ; gh = h0_prev @ w_hh0^T + b_hh0
    gemm_xwT_bias_wmma<<<(3 * HH) / 16, 32, 0, stream>>>(x_bf,     w_ih0, b_ih0, zrow, gi, EE, 3 * HH, 3 * HH);
    gemm_xwT_bias_wmma<<<(3 * HH) / 16, 32, 0, stream>>>(hprev_bf, w_hh0, b_hh0, zrow, gh, HH, 3 * HH, 3 * HH);
    gru_pointwise<<<(BB * HH) / 256, 256, 0, stream>>>(gi, gh, hidden, out_h0, h0_bf);

    // 3) layer 1
    gemm_xwT_bias_wmma<<<(3 * HH) / 16, 32, 0, stream>>>(h0_bf, w_ih1, b_ih1, zrow, gi, HH, 3 * HH, 3 * HH);
    gemm_xwT_bias_wmma<<<(3 * HH) / 16, 32, 0, stream>>>(hprev_bf + (size_t)BB * HH, w_hh1, b_hh1, zrow, gh, HH, 3 * HH, 3 * HH);
    gru_pointwise<<<(BB * HH) / 256, 256, 0, stream>>>(gi, gh, hidden + (size_t)BB * HH, out_h1, h1_bf);

    // 4) logits = h1 @ w_out^T + b_out  (streams 131 MB of w_out exactly once)
    gemm_xwT_bias_wmma<<<(VV + 15) / 16, 32, 0, stream>>>(h1_bf, w_out, b_out, zrow, out, HH, VV, VV);

    // 5) log_softmax in place on d_out
    log_softmax_inplace<<<BB, 256, 0, stream>>>(out);
}
